// PointerNet_56298431316391
// MI455X (gfx1250) — matrix operand order repaired and checked by reading, not compiled, not run on hardware
//
#include <hip/hip_runtime.h>
#include <stdint.h>
#include <stddef.h>

// ---------------- types ----------------
typedef __attribute__((ext_vector_type(16))) __bf16         v16bf;
typedef __attribute__((ext_vector_type(8)))  float          v8f;
typedef __attribute__((ext_vector_type(4)))  unsigned int   u32x4;
typedef __attribute__((ext_vector_type(4)))  unsigned short u16x4;
typedef int v4i_g __attribute__((vector_size(16)));   // matches async-LDS builtin param

#define BATCH   8192
#define DIM     1024
#define BM      32              // rows per workgroup
#define NC      128             // N-chunk width
#define NCHUNK  (DIM / NC)      // 8
#define KSTEP   32
#define KITER   (DIM / KSTEP)   // 32
#define PITCH   (NC + 4)        // 132 floats: bank-conflict-free for half-wave row split
#define SBPITCH 40              // ushorts per sB row (80B, conflict-free, 16B aligned)

#if defined(__has_builtin)
#if __has_builtin(__builtin_amdgcn_global_load_async_to_lds_b128)
#define HAVE_ASYNC_LDS 1
#endif
#endif

__device__ __forceinline__ unsigned short f2bf(float f) {
  unsigned int u = __float_as_uint(f);
  u += 0x7FFFu + ((u >> 16) & 1u);        // round-to-nearest-even
  return (unsigned short)(u >> 16);
}

__device__ __forceinline__ u16x4 cvt4(float4 f) {
  u16x4 r;
  r[0] = f2bf(f.x); r[1] = f2bf(f.y); r[2] = f2bf(f.z); r[3] = f2bf(f.w);
  return r;
}

// 16B async (or synchronous fallback) copy global -> LDS
__device__ __forceinline__ void copy16(const unsigned short* gsrc, unsigned short* ldst) {
#if HAVE_ASYNC_LDS
  __builtin_amdgcn_global_load_async_to_lds_b128((v4i_g*)gsrc, (v4i_g*)ldst, 0, 0);
#else
  *(u32x4*)ldst = *(const u32x4*)gsrc;
#endif
}

__device__ __forceinline__ void async_fence() {
#if HAVE_ASYNC_LDS
#if __has_builtin(__builtin_amdgcn_s_wait_asynccnt)
  __builtin_amdgcn_s_wait_asynccnt(0);
#else
  asm volatile("s_wait_asynccnt 0x0" ::: "memory");
#endif
#endif
}

// ---------------- kernel 1: activations f32 -> bf16 ----------------
__global__ void convert_acts(const float* __restrict__ P,
                             const float* __restrict__ C1, const float* __restrict__ C2,
                             const float* __restrict__ C3, const float* __restrict__ C4,
                             unsigned short* __restrict__ Pb,
                             unsigned short* __restrict__ Cb) {
  const size_t N = (size_t)BATCH * DIM;
  const size_t q = (size_t)blockIdx.x * blockDim.x + threadIdx.x;   // quad index
  if (q >= N / 4) return;
  ((u16x4*)Pb)[q]           = cvt4(((const float4*)P )[q]);
  ((u16x4*)(Cb + 0 * N))[q] = cvt4(((const float4*)C1)[q]);
  ((u16x4*)(Cb + 1 * N))[q] = cvt4(((const float4*)C2)[q]);
  ((u16x4*)(Cb + 2 * N))[q] = cvt4(((const float4*)C3)[q]);
  ((u16x4*)(Cb + 3 * N))[q] = cvt4(((const float4*)C4)[q]);
}

// ---------------- kernel 2: weights f32 [K][N] -> bf16 [N][K] ----------------
__global__ void convert_wt(const float* __restrict__ GwP,
                           const float* __restrict__ GwC,
                           const float* __restrict__ OwW,
                           unsigned short* __restrict__ GpT,
                           unsigned short* __restrict__ GcT,
                           unsigned short* __restrict__ OwT) {
  const int o = blockIdx.x * blockDim.x + threadIdx.x;   // output index, N-major
  if (o >= DIM * DIM) return;
  const int n = o >> 10, k = o & (DIM - 1);
  const int src = k * DIM + n;
  GpT[o] = f2bf(GwP[src]);
  GcT[o] = f2bf(GwC[src]);
  OwT[o] = f2bf(OwW[src]);
}

// ---------------- kernel 3: fused 6-GEMM + gated-dot epilogue ----------------
__global__ __launch_bounds__(256)
void pointer_net_main(const unsigned short* __restrict__ Pb,
                      const unsigned short* __restrict__ Cb,     // [4][BATCH*DIM]
                      const unsigned short* __restrict__ GpT,
                      const unsigned short* __restrict__ GcT,
                      const unsigned short* __restrict__ OwT,
                      const float* __restrict__ C1f, const float* __restrict__ C2f,
                      const float* __restrict__ C3f, const float* __restrict__ C4f,
                      const float* __restrict__ GateBias,
                      const float* __restrict__ OutBias,
                      float* __restrict__ Out) {
  __shared__ float          sWP[BM * PITCH];     // WP tile   (16,896 B)
  __shared__ float          sWO[BM * PITCH];     // WPout tile(16,896 B)
  __shared__ unsigned short sB0[NC * SBPITCH];   // B tile: GpT (ph1) / GcT (ph2)
  __shared__ unsigned short sB1[NC * SBPITCH];   // B tile: OwT (ph1)
  __shared__ float          sGB[DIM];
  __shared__ float          sOB[DIM];
  __shared__ float          sRed[4][BM][4];

  const int tid   = threadIdx.x;
  const int lane  = tid & 31;
  const int wave  = tid >> 5;      // 0..7
  const int strip = wave & 1;      // which 16-row M strip
  const int ng    = wave >> 1;     // 0..3: pair of 16-col N tiles
  const int lh    = lane >> 4;     // half-wave
  const int l16   = lane & 15;
  const int row0  = blockIdx.x * BM;
  const size_t NP = (size_t)BATCH * DIM;

  for (int i = tid; i < DIM; i += 256) { sGB[i] = GateBias[i]; sOB[i] = OutBias[i]; }

  const float* Cf[4] = {C1f, C2f, C3f, C4f};
  float partial[4][8] = {};

  const int arow = row0 + strip * 16 + l16;     // this lane's A-matrix row (M)
  // per-lane fragment column offsets within a staged B tile
  const int nloc0 = (ng * 2 + 0) * 16 + l16;
  const int nloc1 = (ng * 2 + 1) * 16 + l16;

#pragma unroll 1
  for (int ch = 0; ch < NCHUNK; ++ch) {
    const int nb = ch * NC;                     // global col base of this chunk

    // ====== phase 1: WP and WPout tiles (shared A = P) ======
    v8f aWP0 = {}, aWP1 = {}, aWO0 = {}, aWO1 = {};
#pragma unroll 1
    for (int kb = 0; kb < KITER; ++kb) {
      __syncthreads();
      {
        // stage two B tiles (GpT + OwT): thread t -> (weight t&1, row t>>1), 64B each
        const int wsel = tid & 1, r = tid >> 1;
        const unsigned short* Wt = wsel ? OwT : GpT;
        unsigned short* dst = (wsel ? sB1 : sB0) + r * SBPITCH;
        const unsigned short* src = Wt + (size_t)(nb + r) * DIM + kb * KSTEP;
#pragma unroll
        for (int j = 0; j < 4; ++j) copy16(src + j * 8, dst + j * 8);
      }
      async_fence();
      __syncthreads();

      union { v16bf v; u32x4 q[2]; } af;
      const u32x4* ap = (const u32x4*)(Pb + (size_t)arow * DIM + kb * KSTEP + lh * 16);
      af.q[0] = ap[0];
      af.q[1] = ap[1];
      if (kb + 1 < KITER)
        __builtin_prefetch(Pb + (size_t)arow * DIM + (kb + 1) * KSTEP, 0, 3);

      union { v16bf v; u32x4 q[2]; } bP0, bP1, bO0, bO1;
      {
        const u32x4* p0 = (const u32x4*)(sB0 + nloc0 * SBPITCH + lh * 16);
        const u32x4* p1 = (const u32x4*)(sB0 + nloc1 * SBPITCH + lh * 16);
        const u32x4* o0 = (const u32x4*)(sB1 + nloc0 * SBPITCH + lh * 16);
        const u32x4* o1 = (const u32x4*)(sB1 + nloc1 * SBPITCH + lh * 16);
        bP0.q[0] = p0[0]; bP0.q[1] = p0[1];
        bP1.q[0] = p1[0]; bP1.q[1] = p1[1];
        bO0.q[0] = o0[0]; bO0.q[1] = o0[1];
        bO1.q[0] = o1[0]; bO1.q[1] = o1[1];
      }
      // one A fragment feeds 4 independent WMMAs
      aWP0 = __builtin_amdgcn_wmma_f32_16x16x32_bf16(false, af.v, false, bP0.v, (short)0, aWP0, false, false);
      aWP1 = __builtin_amdgcn_wmma_f32_16x16x32_bf16(false, af.v, false, bP1.v, (short)0, aWP1, false, false);
      aWO0 = __builtin_amdgcn_wmma_f32_16x16x32_bf16(false, af.v, false, bO0.v, (short)0, aWO0, false, false);
      aWO1 = __builtin_amdgcn_wmma_f32_16x16x32_bf16(false, af.v, false, bO1.v, (short)0, aWO1, false, false);
    }
    // park WP / WPout tiles in LDS for the epilogue
#pragma unroll
    for (int r = 0; r < 8; ++r) {
      const int m = strip * 16 + lh * 8 + r;
      sWP[m * PITCH + nloc0] = aWP0[r];
      sWP[m * PITCH + nloc1] = aWP1[r];
      sWO[m * PITCH + nloc0] = aWO0[r];
      sWO[m * PITCH + nloc1] = aWO1[r];
    }

    // ====== phase 2: 4 WC GEMMs (shared B = GateWeight_C) ======
    v8f aC[4][2] = {};
#pragma unroll 1
    for (int kb = 0; kb < KITER; ++kb) {
      __syncthreads();
      {
        // stage one B tile (GcT): thread t -> (row t>>1, half t&1), 32B each
        const int r = tid >> 1, half = tid & 1;
        unsigned short* dst = sB0 + r * SBPITCH + half * 16;
        const unsigned short* src = GcT + (size_t)(nb + r) * DIM + kb * KSTEP + half * 16;
        copy16(src, dst);
        copy16(src + 8, dst + 8);
      }
      async_fence();
      __syncthreads();

      // 4 A fragments (one per choice branch)
      union { v16bf v; u32x4 q[2]; } af[4];
#pragma unroll
      for (int k = 0; k < 4; ++k) {
        const u32x4* ap = (const u32x4*)(Cb + (size_t)k * NP + (size_t)arow * DIM + kb * KSTEP + lh * 16);
        af[k].q[0] = ap[0];
        af[k].q[1] = ap[1];
        if (kb + 1 < KITER)
          __builtin_prefetch(Cb + (size_t)k * NP + (size_t)arow * DIM + (kb + 1) * KSTEP, 0, 3);
      }
      // 2 B fragments, each reused by 4 WMMAs from registers
      union { v16bf v; u32x4 q[2]; } bf0, bf1;
      {
        const u32x4* b0 = (const u32x4*)(sB0 + nloc0 * SBPITCH + lh * 16);
        const u32x4* b1 = (const u32x4*)(sB0 + nloc1 * SBPITCH + lh * 16);
        bf0.q[0] = b0[0]; bf0.q[1] = b0[1];
        bf1.q[0] = b1[0]; bf1.q[1] = b1[1];
      }
#pragma unroll
      for (int k = 0; k < 4; ++k) {
        aC[k][0] = __builtin_amdgcn_wmma_f32_16x16x32_bf16(false, af[k].v, false, bf0.v, (short)0, aC[k][0], false, false);
        aC[k][1] = __builtin_amdgcn_wmma_f32_16x16x32_bf16(false, af[k].v, false, bf1.v, (short)0, aC[k][1], false, false);
      }
    }

    // ====== fused epilogue: partial[k] += C_k * sigmoid(WP+WC+gb) * (WPout+ob) ======
#pragma unroll
    for (int k = 0; k < 4; ++k) {
      const float* Ck = Cf[k];
#pragma unroll
      for (int nt = 0; nt < 2; ++nt) {
        const int cl = nt ? nloc1 : nloc0;
        const int cg = nb + cl;
        const float gb = sGB[cg];
        const float ob = sOB[cg];
#pragma unroll
        for (int r = 0; r < 8; ++r) {
          const int m = strip * 16 + lh * 8 + r;
          const float wp = sWP[m * PITCH + cl];
          const float wo = sWO[m * PITCH + cl];
          const float g  = 1.0f / (1.0f + __expf(-(wp + aC[k][nt][r] + gb)));
          const float cv = Ck[(size_t)(row0 + m) * DIM + cg];
          partial[k][r] += cv * g * (wo + ob);
        }
      }
    }
  } // chunk

  // reduce over the 16 lanes of each half-wave (cols), keep rows per half
#pragma unroll
  for (int k = 0; k < 4; ++k)
#pragma unroll
    for (int r = 0; r < 8; ++r) {
      float v = partial[k][r];
      v += __shfl_xor(v, 1, 32);
      v += __shfl_xor(v, 2, 32);
      v += __shfl_xor(v, 4, 32);
      v += __shfl_xor(v, 8, 32);
      partial[k][r] = v;
    }
  if (l16 == 0) {
#pragma unroll
    for (int k = 0; k < 4; ++k)
#pragma unroll
      for (int r = 0; r < 8; ++r) {
        const int m = strip * 16 + lh * 8 + r;
        sRed[ng][m][k] = partial[k][r];
      }
  }
  __syncthreads();
  if (tid < BM * 4) {
    const int m = tid >> 2, k = tid & 3;
    const float v = sRed[0][m][k] + sRed[1][m][k] + sRed[2][m][k] + sRed[3][m][k];
    Out[(size_t)(row0 + m) * 4 + k] = v;
  }
}

// ---------------- host ----------------
extern "C" void kernel_launch(void* const* d_in, const int* in_sizes, int n_in,
                              void* d_out, int out_size, void* d_ws, size_t ws_size,
                              hipStream_t stream) {
  (void)in_sizes; (void)n_in; (void)out_size; (void)ws_size;
  const float* P   = (const float*)d_in[0];
  const float* C1  = (const float*)d_in[1];
  const float* C2  = (const float*)d_in[2];
  const float* C3  = (const float*)d_in[3];
  const float* C4  = (const float*)d_in[4];
  const float* GwP = (const float*)d_in[5];
  const float* GwC = (const float*)d_in[6];
  const float* GB  = (const float*)d_in[7];
  const float* OwW = (const float*)d_in[8];
  const float* OB  = (const float*)d_in[9];
  float* out = (float*)d_out;

  const size_t N = (size_t)BATCH * DIM;          // 8M elements
  char* ws = (char*)d_ws;
  unsigned short* Pb  = (unsigned short*)ws;                 ws += N * 2;
  unsigned short* Cb  = (unsigned short*)ws;                 ws += 4 * N * 2;
  unsigned short* GpT = (unsigned short*)ws;                 ws += (size_t)DIM * DIM * 2;
  unsigned short* GcT = (unsigned short*)ws;                 ws += (size_t)DIM * DIM * 2;
  unsigned short* OwT = (unsigned short*)ws;

  convert_acts<<<dim3((unsigned)(N / 4 / 256)), dim3(256), 0, stream>>>(P, C1, C2, C3, C4, Pb, Cb);
  convert_wt<<<dim3(DIM * DIM / 256), dim3(256), 0, stream>>>(GwP, GwC, OwW, GpT, GcT, OwT);
  pointer_net_main<<<dim3(BATCH / BM), dim3(256), 0, stream>>>(
      Pb, Cb, GpT, GcT, OwT, C1, C2, C3, C4, GB, OB, out);
}
